// Network_21131239096982
// MI455X (gfx1250) — compile-verified, matrix operand order checked by reading, not run on hardware
//
#include <hip/hip_runtime.h>
#include <hip/hip_bf16.h>
#include <math.h>

// ---------------- problem constants ----------------
#define BB   4
#define NN   128
#define EMBD 32
#define MULD 32
#define NBD  10
#define HIDD 128
#define MIDD 256
#define OUTD 128
#define PAIRS (BB*NN*NN)          // 65536
#define KSLICES 16                // split-K slices for the contraction

// ---------------- workspace layout (bytes) ----------------
#define OFF_BASIS 0x00000000ull   // f32 [P,10]            2.62 MB
#define OFF_H1    0x00300000ull   // f16 [P,128]           16.78 MB
#define OFF_H2    0x01400000ull   // f16 [P,128]           16.78 MB
#define OFF_GT    0x02500000ull   // f16 [B,32,16384]      4.19 MB
#define OFF_W2T   0x02A00000ull   // f16 [4,128,128]       0.13 MB
#define OFF_FA    0x02A40000ull   // f32 [B,N,32]          64 KB
#define OFF_FB    0x02A50000ull   // f32 [B,N,32]          64 KB
#define OFF_PART  0x02A60000ull   // f32 [B,16,128,32]     1 MB
#define OFF_FH1   0x02B60000ull   // f32 [B*N,256]         512 KB
#define OFF_FH2   0x02BE0000ull   // f32 [B*N,128]         256 KB

typedef __attribute__((ext_vector_type(16))) _Float16 v16h;
typedef __attribute__((ext_vector_type(8)))  _Float16 v8h;
typedef __attribute__((ext_vector_type(8)))  float    v8f;

#if __has_builtin(__builtin_amdgcn_global_load_async_to_lds_b128) && \
    __has_builtin(__builtin_amdgcn_s_wait_asynccnt)
#define HAVE_ASYNC_LDS 1
typedef int v4i __attribute__((vector_size(4 * sizeof(int))));
typedef __attribute__((address_space(1))) v4i* gvec_ptr;   // global v4i*
typedef __attribute__((address_space(3))) v4i* lvec_ptr;   // LDS v4i*
#else
#define HAVE_ASYNC_LDS 0
#endif

// fast softplus(5x)/5 family: v_exp_f32 + v_log_f32 (TRANS pipe, co-executes
// with WMMA) instead of the ~40-op precise log1pf expansion.
__device__ __forceinline__ float softplus5(float x) {
  float z = 5.0f * x;
  float sp = __logf(1.0f + __expf(z));
  return (z > 20.0f) ? z : sp;
}
__device__ __forceinline__ float act_ssp(float x) {   // (softplus(5x)-ln2)/5
  return (softplus5(x) - 0.69314718055994531f) * 0.2f;
}
__device__ __forceinline__ float act_sp(float x) {    // softplus(5x)/5
  return softplus5(x) * 0.2f;
}

// Load one A/B fragment for V_WMMA_F32_16X16X32_F16.
// rowptr = base of this lane's row (A: M-row, B: N-column in K-major storage),
// koff already includes (lane>>4)*8 shift. Two 16B loads -> *_load_b128.
__device__ __forceinline__ v16h load_frag(const _Float16* rowptr, int koff) {
  v8h lo = *(const v8h*)(rowptr + koff);
  v8h hi = *(const v8h*)(rowptr + koff + 16);
  v16h r;
#pragma unroll
  for (int i = 0; i < 8; ++i) { r[i] = lo[i]; r[i + 8] = hi[i]; }
  return r;
}

// ---------------- tiny prep kernels ----------------

// w2T[l][n][k] = (f16) rw2[l][k][n]  (K-major per output column -> contiguous B loads)
__global__ void k_w2t(const float* __restrict__ rw2, _Float16* __restrict__ w2T) {
  int id = blockIdx.x * blockDim.x + threadIdx.x;   // 4*128*128
  int l = id >> 14, rem = id & 16383, n = rem >> 7, k = rem & 127;
  w2T[(size_t)l * 16384 + n * 128 + k] = (_Float16)rw2[(size_t)l * 16384 + k * 128 + n];
}

// cosine radial basis per pair
__global__ void k_basis(const float* __restrict__ geo, float* __restrict__ basis) {
  int p = blockIdx.x * blockDim.x + threadIdx.x;    // 65536
  int b = p >> 14, rem = p & 16383, x = rem >> 7, y = rem & 127;
  float dx = geo[(b * NN + y) * 3 + 0] - geo[(b * NN + x) * 3 + 0];
  float dy = geo[(b * NN + y) * 3 + 1] - geo[(b * NN + x) * 3 + 1];
  float dz = geo[(b * NN + y) * 3 + 2] - geo[(b * NN + x) * 3 + 2];
  float r = sqrtf(dx * dx + dy * dy + dz * dz);
  const float step = 10.0f / 9.0f, inv_step = 0.9f;
#pragma unroll
  for (int k = 0; k < NBD; ++k) {
    float xx = (r - (float)k * step) * inv_step;
    float v = 0.0f;
    if (fabsf(xx) < 1.0f) { float c = cosf(1.57079632679f * xx); v = c * c; }
    basis[(size_t)p * NBD + k] = v;
  }
}

// f0 = emb[features]
__global__ void k_f0(const int* __restrict__ feat, const float* __restrict__ emb,
                     float* __restrict__ f) {
  int id = blockIdx.x * blockDim.x + threadIdx.x;   // 16384
  int a = id >> 5, i = id & 31;                     // a = b*128+n
  f[id] = emb[feat[a] * EMBD + i];
}

// ---------------- per-layer kernels ----------------

// H1 = ssp(basis @ w1 / sqrt(10))  -> f16 [P,128]
__global__ void __launch_bounds__(256) k_h1(const float* __restrict__ basis,
                                            const float* __restrict__ w1,
                                            _Float16* __restrict__ H1) {
  __shared__ float w1s[NBD * HIDD];                 // 5 KB
  for (int i = threadIdx.x; i < NBD * HIDD; i += 256) w1s[i] = w1[i];
  __syncthreads();
  int p = blockIdx.x * 256 + threadIdx.x;           // 256 blocks
  float bk[NBD];
#pragma unroll
  for (int k = 0; k < NBD; ++k) bk[k] = basis[(size_t)p * NBD + k];
  const float s = 0.31622776601f;                   // 1/sqrt(10)
  for (int h = 0; h < HIDD; ++h) {
    float acc = 0.0f;
#pragma unroll
    for (int k = 0; k < NBD; ++k) acc += bk[k] * w1s[k * HIDD + h];
    H1[(size_t)p * HIDD + h] = (_Float16)act_ssp(acc * s);
  }
}

// H2 = ssp(H1 @ w2 / sqrt(128)) : [65536,128] x [128,128] WMMA GEMM.
// block = 8 waves, tile = 16 rows x 128 cols; the shared 16x128 A tile is
// staged once into LDS (async global->LDS when available), each wave takes
// one 16-col B tile directly from global (w2T is L2-resident).
__global__ void __launch_bounds__(256) k_gemm_h2(const _Float16* __restrict__ H1,
                                                 const _Float16* __restrict__ w2T,
                                                 _Float16* __restrict__ H2) {
  const int lane = threadIdx.x & 31;
  const int wave = threadIdx.x >> 5;        // col tile 0..7
  const int rowBase = blockIdx.x * 16;      // 4096 blocks
  const int colBase = wave * 16;
  const int laneLo = lane & 15;
  const int laneHi = lane >> 4;

  __shared__ _Float16 As[16 * 128];         // 4 KB
  {
    // 2048 halves = 256 x 16B chunks; thread t copies chunk t.
    int t = threadIdx.x;
    int row = t >> 4, off = (t & 15) * 8;
    const _Float16* g = H1 + (size_t)(rowBase + row) * 128 + off;
    _Float16* l = &As[row * 128 + off];
#if HAVE_ASYNC_LDS
    __builtin_amdgcn_global_load_async_to_lds_b128((gvec_ptr)g, (lvec_ptr)l, 0, 0);
    __builtin_amdgcn_s_wait_asynccnt(0);
#else
    *(v8h*)l = *(const v8h*)g;
#endif
    __syncthreads();
  }

  const _Float16* aRow = &As[laneLo * 128];
  const _Float16* bRow = w2T + (size_t)(colBase + laneLo) * 128;
  v8f c = {};
#pragma unroll
  for (int ks = 0; ks < 4; ++ks) {
    int k0 = ks * 32 + laneHi * 8;
    v16h a = load_frag(aRow, k0);           // ds_load_b128 (broadcast across waves)
    v16h b = load_frag(bRow, k0);           // global_load_b128
    c = __builtin_amdgcn_wmma_f32_16x16x32_f16(false, a, false, b, (short)0, c, false, false);
  }
  const float s = 0.08838834765f;           // 1/sqrt(128)
#pragma unroll
  for (int r = 0; r < 8; ++r) {
    int row = rowBase + r + laneHi * 8;
    int col = colBase + laneLo;
    H2[(size_t)row * 128 + col] = (_Float16)act_ssp(c[r] * s);
  }
}

// Gt[b][i][y*128+h] = sum_j w3[h, i*32+j] * f[b,y,j]*mask[b,y]/sqrt(N)
__global__ void __launch_bounds__(256) k_G(const float* __restrict__ f,
                                           const float* __restrict__ mask,
                                           const float* __restrict__ w3,
                                           _Float16* __restrict__ Gt) {
  int by = blockIdx.x;                      // 512 blocks = b*128+y
  int b = by >> 7, y = by & 127;
  __shared__ float fj[32];
  if (threadIdx.x < 32)
    fj[threadIdx.x] = f[(size_t)by * 32 + threadIdx.x] * mask[by] * 0.08838834765f;
  __syncthreads();
  for (int idx = threadIdx.x; idx < HIDD * MULD; idx += 256) {
    int h = idx >> 5, i = idx & 31;
    const float* w = w3 + (size_t)h * 1024 + i * 32;
    float acc = 0.0f;
#pragma unroll
    for (int j = 0; j < 32; ++j) acc += w[j] * fj[j];
    Gt[((size_t)(b * 32 + i)) * 16384 + y * 128 + h] = (_Float16)acc;
  }
}

// out_part[b][s][x][i] = sum over K-slice of H2[b,x,(y,h)] * Gt[b,i,(y,h)]
// grid: B * 16 slices * 2 n-tiles blocks; 8 waves = 8 m-tiles; K per slice = 1024.
// The 16-col x 1024-k Gt slice (32 KB) is shared by all 8 waves -> stage it in
// LDS once per block via async global->LDS, then feed B fragments from LDS.
__global__ void __launch_bounds__(256) k_contract(const _Float16* __restrict__ H2,
                                                  const _Float16* __restrict__ Gt,
                                                  float* __restrict__ part) {
  int bid = blockIdx.x;
  int nt = bid & 1;  bid >>= 1;
  int s  = bid & 15; bid >>= 4;
  int b  = bid;
  const int lane = threadIdx.x & 31;
  const int mt = threadIdx.x >> 5;
  const int laneLo = lane & 15, laneHi = lane >> 4;
  const int x = mt * 16 + laneLo;
  const _Float16* aRow = H2 + ((size_t)(b * 128 + x)) * 16384 + s * 1024;

  __shared__ _Float16 Bs[16 * 1024];        // 32 KB: [col][k]
  {
    const _Float16* src0 = Gt + ((size_t)(b * 32 + nt * 16)) * 16384 + s * 1024;
    // 16 cols x 1024 halves = 2048 x 16B chunks; 8 chunks per thread.
    for (int cchunk = threadIdx.x; cchunk < 2048; cchunk += 256) {
      int col = cchunk >> 7, off = (cchunk & 127) * 8;
      const _Float16* g = src0 + (size_t)col * 16384 + off;
      _Float16* l = &Bs[col * 1024 + off];
#if HAVE_ASYNC_LDS
      __builtin_amdgcn_global_load_async_to_lds_b128((gvec_ptr)g, (lvec_ptr)l, 0, 0);
#else
      *(v8h*)l = *(const v8h*)g;
#endif
    }
#if HAVE_ASYNC_LDS
    __builtin_amdgcn_s_wait_asynccnt(0);
#endif
    __syncthreads();
  }

  const _Float16* bCol = &Bs[laneLo * 1024];
  v8f c = {};
  for (int ks = 0; ks < 32; ++ks) {
    int k0 = ks * 32 + laneHi * 8;
    v16h a  = load_frag(aRow, k0);          // global_load_b128
    v16h bf = load_frag(bCol, k0);          // ds_load_b128
    c = __builtin_amdgcn_wmma_f32_16x16x32_f16(false, a, false, bf, (short)0, c, false, false);
  }
#pragma unroll
  for (int r = 0; r < 8; ++r) {
    int row = mt * 16 + r + laneHi * 8;
    int col = nt * 16 + laneLo;
    part[(((size_t)(b * 16 + s) * 128) + row) * 32 + col] = c[r];
  }
}

// fnext = sp( (Y0/sqrt(HID)) * sum_s part ) * mask
__global__ void k_reduce(const float* __restrict__ part, const float* __restrict__ mask,
                         float* __restrict__ fn) {
  int id = blockIdx.x * blockDim.x + threadIdx.x;   // 16384
  int b = id >> 12, rem = id & 4095, x = rem >> 5, i = rem & 31;
  float acc = 0.0f;
#pragma unroll
  for (int s = 0; s < KSLICES; ++s)
    acc += part[(((size_t)(b * KSLICES + s) * 128) + x) * 32 + i];
  const float sc = 0.28209479177f * 0.08838834765f; // Y0/sqrt(128)
  fn[id] = act_sp(acc * sc) * mask[b * NN + x];
}

// ---------------- final MLP / BN / pool ----------------

// generic FC: out[row,c] = bias[c] + sum_j in[row,j]*W[j,c]
__global__ void k_fc(const float* __restrict__ in, const float* __restrict__ W,
                     const float* __restrict__ bias, float* __restrict__ out,
                     int IN, int OUT) {
  int id = blockIdx.x * blockDim.x + threadIdx.x;
  int row = id / OUT, c = id - row * OUT;
  float acc = bias[c];
  for (int j = 0; j < IN; ++j) acc += in[(size_t)row * IN + j] * W[(size_t)j * OUT + c];
  out[id] = acc;
}

// BatchNorm over (batch, feature) per atom channel n, then leaky_relu(0.2)
__global__ void __launch_bounds__(256) k_bn(float* __restrict__ x,
                                            const float* __restrict__ g,
                                            const float* __restrict__ be, int C) {
  int n = blockIdx.x;                       // 128 blocks
  int t = threadIdx.x;
  __shared__ float ssum[256], ssq[256];
  float s = 0.0f, q = 0.0f;
  int total = BB * C;
  for (int idx = t; idx < total; idx += 256) {
    int b = idx / C, c = idx - b * C;
    float v = x[((size_t)(b * NN + n)) * C + c];
    s += v; q += v * v;
  }
  ssum[t] = s; ssq[t] = q; __syncthreads();
  for (int st = 128; st > 0; st >>= 1) {
    if (t < st) { ssum[t] += ssum[t + st]; ssq[t] += ssq[t + st]; }
    __syncthreads();
  }
  float inv = 1.0f / (float)total;
  float m = ssum[0] * inv;
  float v = ssq[0] * inv - m * m;
  float rs = rsqrtf(v + 1e-5f);
  float gg = g[n], bb2 = be[n];
  for (int idx = t; idx < total; idx += 256) {
    int b = idx / C, c = idx - b * C;
    size_t off = ((size_t)(b * NN + n)) * C + c;
    float z = (x[off] - m) * rs * gg + bb2;
    x[off] = (z < 0.0f) ? 0.2f * z : z;
  }
}

// masked mean over atoms
__global__ void k_pool(const float* __restrict__ h, const float* __restrict__ mask,
                       float* __restrict__ out) {
  int t = blockIdx.x * blockDim.x + threadIdx.x;    // 512
  int b = t >> 7, o = t & 127;
  float acc = 0.0f, ms = 0.0f;
  for (int n = 0; n < NN; ++n) {
    float mk = mask[b * NN + n];
    acc += h[((size_t)(b * NN + n)) * OUTD + o] * mk;
    ms += mk;
  }
  out[t] = acc / ms;
}

// ---------------- host launch ----------------
extern "C" void kernel_launch(void* const* d_in, const int* in_sizes, int n_in,
                              void* d_out, int out_size, void* d_ws, size_t ws_size,
                              hipStream_t stream) {
  const int*   features = (const int*)  d_in[0];
  const float* geometry = (const float*)d_in[1];
  const float* mask     = (const float*)d_in[2];
  const float* emb      = (const float*)d_in[3];
  const float* rw1      = (const float*)d_in[4];
  const float* rw2      = (const float*)d_in[5];
  const float* rw3      = (const float*)d_in[6];
  const float* W1       = (const float*)d_in[7];
  const float* b1       = (const float*)d_in[8];
  const float* g1       = (const float*)d_in[9];
  const float* be1      = (const float*)d_in[10];
  const float* W2       = (const float*)d_in[11];
  const float* b2       = (const float*)d_in[12];
  const float* g2       = (const float*)d_in[13];
  const float* be2      = (const float*)d_in[14];
  float* out = (float*)d_out;

  char* ws = (char*)d_ws;
  float*    basis = (float*)   (ws + OFF_BASIS);
  _Float16* H1    = (_Float16*)(ws + OFF_H1);
  _Float16* H2    = (_Float16*)(ws + OFF_H2);
  _Float16* Gt    = (_Float16*)(ws + OFF_GT);
  _Float16* w2T   = (_Float16*)(ws + OFF_W2T);
  float*    fA    = (float*)   (ws + OFF_FA);
  float*    fB    = (float*)   (ws + OFF_FB);
  float*    part  = (float*)   (ws + OFF_PART);
  float*    fh1   = (float*)   (ws + OFF_FH1);
  float*    fh2   = (float*)   (ws + OFF_FH2);

  k_w2t  <<<256, 256, 0, stream>>>(rw2, w2T);
  k_basis<<<PAIRS / 256, 256, 0, stream>>>(geometry, basis);
  k_f0   <<<64, 256, 0, stream>>>(features, emb, fA);

  float* fcur = fA;
  float* fnxt = fB;
  for (int l = 0; l < 4; ++l) {
    k_h1      <<<256, 256, 0, stream>>>(basis, rw1 + (size_t)l * NBD * HIDD, H1);
    k_gemm_h2 <<<PAIRS / 16, 256, 0, stream>>>(H1, w2T + (size_t)l * 16384, H2);
    k_G       <<<BB * NN, 256, 0, stream>>>(fcur, mask, rw3 + (size_t)l * HIDD * 1024, Gt);
    k_contract<<<BB * KSLICES * 2, 256, 0, stream>>>(H2, Gt, part);
    k_reduce  <<<64, 256, 0, stream>>>(part, mask, fnxt);
    float* tmp = fcur; fcur = fnxt; fnxt = tmp;
  }

  k_fc  <<<(BB * NN * MIDD) / 256, 256, 0, stream>>>(fcur, W1, b1, fh1, EMBD, MIDD);
  k_bn  <<<NN, 256, 0, stream>>>(fh1, g1, be1, MIDD);
  k_fc  <<<(BB * NN * OUTD) / 256, 256, 0, stream>>>(fh1, W2, b2, fh2, MIDD, OUTD);
  k_bn  <<<NN, 256, 0, stream>>>(fh2, g2, be2, OUTD);
  k_pool<<<2, 256, 0, stream>>>(fh2, mask, out);
  (void)in_sizes; (void)n_in; (void)out_size; (void)ws_size;
}
